// BlockLevelRouter_64355789963655
// MI455X (gfx1250) — matrix operand order repaired and checked by reading, not compiled, not run on hardware
//
#include <hip/hip_runtime.h>
#include <hip/hip_bf16.h>

// ---------------------------------------------------------------------------
// BlockLevelRouter for MI455X (gfx1250): bf16 WMMA GEMMs with fused epilogues
// B=2048, NB=101, H=256, E=656, NCT=32, NL=2.  M = B*NB = 206848 = 1616*128.
// FFN layers fully fused per 128-row block: h (128x512) lives in LDS (CDNA5
// 320KB WGP LDS), eliminating 1.7 GB of HBM traffic for the h intermediate.
// ---------------------------------------------------------------------------

#define B_   2048
#define NB_  101
#define H_   256
#define E_   656
#define M_   (B_ * NB_)          // 206848
#define LN_EPS 1e-5f

typedef __attribute__((ext_vector_type(16))) __bf16 v16bf;
typedef __attribute__((ext_vector_type(8)))  float  v8f;

// epilogue modes
#define EPI_ELU  0
#define EPI_GATE 1

// LDS stride for staging tiles: 34 half-words (17 dwords)
#define LDH 34
#define LDW 17
// fused-FFN h tile: 128 rows x 512 cols bf16, padded stride 514 halves
#define HLDH 514
#define HLDW 257
#define HS_DW   (128 * HLDW)     // 32896 dwords
#define STG_DW  (256 * LDW)      // 4352 dwords (A+B staging / W2 staging)
#define FFN_LDS ((HS_DW + STG_DW) * 4)   // 148992 bytes

__device__ __forceinline__ unsigned int pk_bf16(float lo, float hi) {
    union { __bf16 h[2]; unsigned int u; } c;
    c.h[0] = (__bf16)lo; c.h[1] = (__bf16)hi;
    return c.u;
}

__device__ __forceinline__ float elu_f(float v) {
    return v > 0.0f ? v : (__expf(v) - 1.0f);
}

// ---------------------------------------------------------------------------
// Tiled WMMA GEMM: out = epi( A @ W + bias )
//   block tile 128x128, BK=32, 256 threads = 8 waves (4 rowg x 2 colg),
//   each wave 32x64 via 8 accumulators; fp32->bf16 packed while staging.
// ---------------------------------------------------------------------------
template <int EPI, bool SPLIT>
__global__ __launch_bounds__(256)
void gemm_kernel(const float* __restrict__ A0, const float* __restrict__ A1,
                 const float* __restrict__ W,  const float* __restrict__ bias,
                 const float* __restrict__ xres, const float* __restrict__ msg,
                 float* __restrict__ out, int N, int K, int K0)
{
    __shared__ unsigned int As[128 * LDW];
    __shared__ unsigned int Bs[128 * LDW];
    const __bf16* Ash = (const __bf16*)As;
    const __bf16* Bsh = (const __bf16*)Bs;

    const int tid  = threadIdx.x;
    const int wave = tid >> 5;
    const int lane = tid & 31;
    const int half = lane >> 4;
    const int l16  = lane & 15;
    const int rowg = wave >> 1;
    const int colg = wave & 1;

    const int rowBase = blockIdx.x * 128;
    const int colBase = blockIdx.y * 128;

    v8f acc[2][4];
#pragma unroll
    for (int g = 0; g < 2; ++g)
#pragma unroll
        for (int j = 0; j < 4; ++j)
#pragma unroll
            for (int r = 0; r < 8; ++r) acc[g][j][r] = 0.0f;

    for (int kk = 0; kk < K; kk += 32) {
        {   // stage A tile (128 x 32)
            const int r  = tid >> 1;
            const int ks = (tid & 1) * 16;
            const int grow  = rowBase + r;
            const int kglob = kk + ks;
            const float* src;
            if (SPLIT && kglob >= K0)
                src = A1 + (size_t)grow * (size_t)(K - K0) + (kglob - K0);
            else
                src = A0 + (size_t)grow * (size_t)(SPLIT ? K0 : K) + kglob;
            float4 t[4];
#pragma unroll
            for (int i = 0; i < 4; ++i) t[i] = ((const float4*)src)[i];
            const float f[16] = {t[0].x,t[0].y,t[0].z,t[0].w, t[1].x,t[1].y,t[1].z,t[1].w,
                                 t[2].x,t[2].y,t[2].z,t[2].w, t[3].x,t[3].y,t[3].z,t[3].w};
            unsigned int* dst = &As[r * LDW + (ks >> 1)];
#pragma unroll
            for (int i = 0; i < 8; ++i) dst[i] = pk_bf16(f[2*i], f[2*i+1]);
        }
        {   // stage B tile (32 x 128) transposed as Bs[n][k], k-pairs packed
            const int kp = (tid >> 4) << 1;
            const int ns = (tid & 15) * 8;
            const float* w0 = W + (size_t)(kk + kp) * (size_t)N + colBase + ns;
            const float* w1 = w0 + N;
            const float4 x0 = ((const float4*)w0)[0];
            const float4 x1 = ((const float4*)w0)[1];
            const float4 y0 = ((const float4*)w1)[0];
            const float4 y1 = ((const float4*)w1)[1];
            const float f0[8] = {x0.x,x0.y,x0.z,x0.w,x1.x,x1.y,x1.z,x1.w};
            const float f1[8] = {y0.x,y0.y,y0.z,y0.w,y1.x,y1.y,y1.z,y1.w};
#pragma unroll
            for (int i = 0; i < 8; ++i)
                Bs[(ns + i) * LDW + (kp >> 1)] = pk_bf16(f0[i], f1[i]);
        }
        __syncthreads();

        v16bf a[2];
#pragma unroll
        for (int g = 0; g < 2; ++g) {
            const int arow = (rowg << 5) + (g << 4) + l16;
#pragma unroll
            for (int e = 0; e < 16; ++e) {
                const int k = ((e & 8) << 1) + (half << 3) + (e & 7);
                a[g][e] = Ash[arow * LDH + k];
            }
        }
#pragma unroll
        for (int j = 0; j < 4; ++j) {
            v16bf b;
            const int bcol = (colg << 6) + (j << 4) + l16;
#pragma unroll
            for (int e = 0; e < 16; ++e)
                b[e] = Bsh[bcol * LDH + ((e & 8) << 1) + (half << 3) + (e & 7)];
            acc[0][j] = __builtin_amdgcn_wmma_f32_16x16x32_bf16(
                false, a[0], false, b, (short)0, acc[0][j], false, false);
            acc[1][j] = __builtin_amdgcn_wmma_f32_16x16x32_bf16(
                false, a[1], false, b, (short)0, acc[1][j], false, false);
        }
        __syncthreads();
    }

#pragma unroll
    for (int g = 0; g < 2; ++g) {
        const int mrow = rowBase + (rowg << 5) + (g << 4) + (half << 3);
#pragma unroll
        for (int j = 0; j < 4; ++j) {
            const int gcol = colBase + (colg << 6) + (j << 4) + l16;
            const float bb = bias[gcol];
#pragma unroll
            for (int r = 0; r < 8; ++r) {
                const size_t idx = (size_t)(mrow + r) * (size_t)N + gcol;
                float v = acc[g][j][r] + bb;
                if (EPI == EPI_ELU) {
                    v = elu_f(v);
                } else { // EPI_GATE (N == H)
                    const float gt = 1.0f / (1.0f + __expf(-v));
                    v = xres[idx] + gt * msg[idx];
                }
                out[idx] = v;
            }
        }
    }
}

// ---------------------------------------------------------------------------
// Fused pre-norm FFN layer:  x += elu(LN(x)@W1 + b1) @ W2 + b2
// One 128-row block per workgroup. Phase 1 builds the full 128x512 bf16 h
// tile in LDS (4 column chunks); phase 2 streams W2 tiles through LDS and
// reads A-fragments straight from the h tile. No h traffic to HBM.
// ---------------------------------------------------------------------------
__global__ __launch_bounds__(256)
void ffn_kernel(const float* __restrict__ x, const float* __restrict__ stats,
                const float* __restrict__ lng, const float* __restrict__ lnb,
                const float* __restrict__ W1, const float* __restrict__ b1,
                const float* __restrict__ W2, const float* __restrict__ b2,
                float* __restrict__ out)
{
    extern __shared__ unsigned int sh[];
    unsigned int* Hs = sh;                      // 128 x HLDW dwords
    unsigned int* As = sh + HS_DW;              // 128 x LDW
    unsigned int* Bs = As + 128 * LDW;          // 128 x LDW
    unsigned int* B2 = sh + HS_DW;              // 256 x LDW (aliases As+Bs)
    __bf16* Hsh = (__bf16*)Hs;
    const __bf16* Ash = (const __bf16*)As;
    const __bf16* Bsh = (const __bf16*)Bs;
    const __bf16* B2h = (const __bf16*)B2;

    const int tid  = threadIdx.x;
    const int wave = tid >> 5;
    const int lane = tid & 31;
    const int half = lane >> 4;
    const int l16  = lane & 15;
    const int rowg = wave >> 1;
    const int colg = wave & 1;
    const int rowBase = blockIdx.x * 128;

    // ---------------- phase 1: Hs = elu(LN(x) @ W1 + b1) ----------------
#pragma unroll 1
    for (int nc = 0; nc < 4; ++nc) {
        v8f acc[2][4];
#pragma unroll
        for (int g = 0; g < 2; ++g)
#pragma unroll
            for (int j = 0; j < 4; ++j)
#pragma unroll
                for (int r = 0; r < 8; ++r) acc[g][j][r] = 0.0f;

#pragma unroll 1
        for (int kk = 0; kk < H_; kk += 32) {
            {   // stage LN(x) tile (128 x 32)
                const int r  = tid >> 1;
                const int ks = (tid & 1) * 16;
                const int grow  = rowBase + r;
                const int kglob = kk + ks;
                const float* src = x + (size_t)grow * H_ + kglob;
                float4 t[4];
#pragma unroll
                for (int i = 0; i < 4; ++i) t[i] = ((const float4*)src)[i];
                float f[16] = {t[0].x,t[0].y,t[0].z,t[0].w, t[1].x,t[1].y,t[1].z,t[1].w,
                               t[2].x,t[2].y,t[2].z,t[2].w, t[3].x,t[3].y,t[3].z,t[3].w};
                const float mu = stats[2 * grow];
                const float rs = stats[2 * grow + 1];
#pragma unroll
                for (int i = 0; i < 16; ++i)
                    f[i] = (f[i] - mu) * rs * lng[kglob + i] + lnb[kglob + i];
                unsigned int* dst = &As[r * LDW + (ks >> 1)];
#pragma unroll
                for (int i = 0; i < 8; ++i) dst[i] = pk_bf16(f[2*i], f[2*i+1]);
            }
            {   // stage W1 tile (32 x 128 of N=512)
                const int kp = (tid >> 4) << 1;
                const int ns = (tid & 15) * 8;
                const float* w0 = W1 + (size_t)(kk + kp) * (2 * H_) + nc * 128 + ns;
                const float* w1 = w0 + 2 * H_;
                const float4 x0 = ((const float4*)w0)[0];
                const float4 x1 = ((const float4*)w0)[1];
                const float4 y0 = ((const float4*)w1)[0];
                const float4 y1 = ((const float4*)w1)[1];
                const float f0[8] = {x0.x,x0.y,x0.z,x0.w,x1.x,x1.y,x1.z,x1.w};
                const float f1[8] = {y0.x,y0.y,y0.z,y0.w,y1.x,y1.y,y1.z,y1.w};
#pragma unroll
                for (int i = 0; i < 8; ++i)
                    Bs[(ns + i) * LDW + (kp >> 1)] = pk_bf16(f0[i], f1[i]);
            }
            __syncthreads();

            v16bf a[2];
#pragma unroll
            for (int g = 0; g < 2; ++g) {
                const int arow = (rowg << 5) + (g << 4) + l16;
#pragma unroll
                for (int e = 0; e < 16; ++e) {
                    const int k = ((e & 8) << 1) + (half << 3) + (e & 7);
                    a[g][e] = Ash[arow * LDH + k];
                }
            }
#pragma unroll
            for (int j = 0; j < 4; ++j) {
                v16bf b;
                const int bcol = (colg << 6) + (j << 4) + l16;
#pragma unroll
                for (int e = 0; e < 16; ++e)
                    b[e] = Bsh[bcol * LDH + ((e & 8) << 1) + (half << 3) + (e & 7)];
                acc[0][j] = __builtin_amdgcn_wmma_f32_16x16x32_bf16(
                    false, a[0], false, b, (short)0, acc[0][j], false, false);
                acc[1][j] = __builtin_amdgcn_wmma_f32_16x16x32_bf16(
                    false, a[1], false, b, (short)0, acc[1][j], false, false);
            }
            __syncthreads();
        }
        // epilogue -> h tile in LDS (bias + elu, bf16)
#pragma unroll
        for (int g = 0; g < 2; ++g) {
            const int lrow = (rowg << 5) + (g << 4) + (half << 3);
#pragma unroll
            for (int j = 0; j < 4; ++j) {
                const int col = nc * 128 + (colg << 6) + (j << 4) + l16;
                const float bb = b1[col];
#pragma unroll
                for (int r = 0; r < 8; ++r)
                    Hsh[(lrow + r) * HLDH + col] = (__bf16)elu_f(acc[g][j][r] + bb);
            }
        }
    }
    __syncthreads();   // h tile complete before phase-2 fragment reads

    // ---------------- phase 2: out = x + Hs @ W2 + b2 ----------------
    v8f acc2[2][8];
#pragma unroll
    for (int g = 0; g < 2; ++g)
#pragma unroll
        for (int j = 0; j < 8; ++j)
#pragma unroll
            for (int r = 0; r < 8; ++r) acc2[g][j][r] = 0.0f;

#pragma unroll 1
    for (int kk = 0; kk < 2 * H_; kk += 32) {
        {   // stage W2 tile (32 x 256) transposed, k-pairs packed
            const int kp = (tid >> 4) << 1;
            const int ns = (tid & 15) * 16;
            const float* w0 = W2 + (size_t)(kk + kp) * H_ + ns;
            const float* w1 = w0 + H_;
            float f0[16], f1[16];
#pragma unroll
            for (int i = 0; i < 4; ++i) {
                const float4 u = ((const float4*)w0)[i];
                const float4 v = ((const float4*)w1)[i];
                f0[4*i+0]=u.x; f0[4*i+1]=u.y; f0[4*i+2]=u.z; f0[4*i+3]=u.w;
                f1[4*i+0]=v.x; f1[4*i+1]=v.y; f1[4*i+2]=v.z; f1[4*i+3]=v.w;
            }
#pragma unroll
            for (int i = 0; i < 16; ++i)
                B2[(ns + i) * LDW + (kp >> 1)] = pk_bf16(f0[i], f1[i]);
        }
        __syncthreads();

        v16bf a[2];
#pragma unroll
        for (int g = 0; g < 2; ++g) {
            const int arow = (rowg << 5) + (g << 4) + l16;
#pragma unroll
            for (int e = 0; e < 16; ++e) {
                const int k = ((e & 8) << 1) + (half << 3) + (e & 7);
                a[g][e] = Hsh[arow * HLDH + kk + k];
            }
        }
#pragma unroll
        for (int j = 0; j < 8; ++j) {
            v16bf b;
            const int bcol = (colg << 7) + (j << 4) + l16;
#pragma unroll
            for (int e = 0; e < 16; ++e)
                b[e] = B2h[bcol * LDH + ((e & 8) << 1) + (half << 3) + (e & 7)];
            acc2[0][j] = __builtin_amdgcn_wmma_f32_16x16x32_bf16(
                false, a[0], false, b, (short)0, acc2[0][j], false, false);
            acc2[1][j] = __builtin_amdgcn_wmma_f32_16x16x32_bf16(
                false, a[1], false, b, (short)0, acc2[1][j], false, false);
        }
        __syncthreads();
    }

#pragma unroll
    for (int g = 0; g < 2; ++g) {
        const int mrow = rowBase + (rowg << 5) + (g << 4) + (half << 3);
#pragma unroll
        for (int j = 0; j < 8; ++j) {
            const int gcol = (colg << 7) + (j << 4) + l16;
            const float bb = b2[gcol];
#pragma unroll
            for (int r = 0; r < 8; ++r) {
                const size_t idx = (size_t)(mrow + r) * H_ + gcol;
                out[idx] = acc2[g][j][r] + bb + x[idx];
            }
        }
    }
}

// ---------------------------------------------------------------------------
// edge_w[b,e] = sigmoid(edge_structure[e]) * ct_w[cancer_type[b], e] * active
// ---------------------------------------------------------------------------
__global__ __launch_bounds__(256)
void edgew_kernel(const float* __restrict__ es, const float* __restrict__ ctw,
                  const int* __restrict__ ct, const unsigned char* __restrict__ ba,
                  const int* __restrict__ esrc, const int* __restrict__ edst,
                  float* __restrict__ out)
{
    const int idx = blockIdx.x * 256 + threadIdx.x;
    if (idx >= B_ * E_) return;
    const int b = idx / E_;
    const int e = idx - b * E_;
    const float sg = 1.0f / (1.0f + __expf(-es[e]));
    float w = sg * ctw[ct[b] * E_ + e];
    const int s = esrc[e], d = edst[e];
    const float act = (ba[b * NB_ + s] && ba[b * NB_ + d]) ? 1.0f : 0.0f;
    out[idx] = w * act;
}

// ---------------------------------------------------------------------------
// messages: one workgroup per batch; full 101x256 fp32 tile in LDS;
// edges processed wave-parallel with ds_add_f32 atomics.
// ---------------------------------------------------------------------------
__global__ __launch_bounds__(256)
void messages_kernel(const float* __restrict__ hroute, const float* __restrict__ edge_w,
                     const int* __restrict__ esrc, const int* __restrict__ edst,
                     float* __restrict__ msg)
{
    extern __shared__ float smsg[];   // NB_ * H_ floats
    const int b   = blockIdx.x;
    const int tid = threadIdx.x;
    for (int i = tid; i < NB_ * H_; i += 256) smsg[i] = 0.0f;
    __syncthreads();

    const int wid = tid >> 5, lane = tid & 31;
    const float* hb  = hroute + (size_t)b * NB_ * H_;
    const float* ewb = edge_w + (size_t)b * E_;
    for (int e = wid; e < E_; e += 8) {
        const float w = ewb[e];
        const int s = esrc[e], d = edst[e];
        const float* hr = hb + (size_t)s * H_;
        float* mr = smsg + d * H_;
#pragma unroll
        for (int i = 0; i < 8; ++i) {
            const int c = lane + (i << 5);
            atomicAdd(&mr[c], w * hr[c]);
        }
    }
    __syncthreads();
    float* mb = msg + (size_t)b * NB_ * H_;
    for (int i = tid; i < NB_ * H_; i += 256) mb[i] = smsg[i];
}

// ---------------------------------------------------------------------------
// per-row LayerNorm stats: wave32 per row, shfl_xor reduction
// ---------------------------------------------------------------------------
__global__ __launch_bounds__(256)
void ln_stats_kernel(const float* __restrict__ x, float* __restrict__ stats)
{
    const int row  = blockIdx.x * 8 + (threadIdx.x >> 5);
    const int lane = threadIdx.x & 31;
    const float* p = x + (size_t)row * H_;
    float s = 0.0f, s2 = 0.0f;
#pragma unroll
    for (int i = 0; i < 8; ++i) {
        const float v = p[lane + (i << 5)];
        s += v; s2 += v * v;
    }
#pragma unroll
    for (int off = 16; off; off >>= 1) {
        s  += __shfl_xor(s,  off, 32);
        s2 += __shfl_xor(s2, off, 32);
    }
    if (lane == 0) {
        const float mu  = s * (1.0f / H_);
        const float var = s2 * (1.0f / H_) - mu * mu;
        stats[2 * row]     = mu;
        stats[2 * row + 1] = rsqrtf(var + LN_EPS);
    }
}

// ---------------------------------------------------------------------------
extern "C" void kernel_launch(void* const* d_in, const int* in_sizes, int n_in,
                              void* d_out, int out_size, void* d_ws, size_t ws_size,
                              hipStream_t stream)
{
    (void)in_sizes; (void)n_in; (void)out_size; (void)ws_size;

    const float*         bt   = (const float*)d_in[0];
    const int*           ct   = (const int*)d_in[1];
    const unsigned char* ba   = (const unsigned char*)d_in[2];
    const int*           esrc = (const int*)d_in[3];
    const int*           edst = (const int*)d_in[4];
    const float*         es   = (const float*)d_in[5];
    const float*         ctw  = (const float*)d_in[6];
    const float*         pW   = (const float*)d_in[7];
    const float*         pb   = (const float*)d_in[8];
    const float*         gW   = (const float*)d_in[9];
    const float*         gb   = (const float*)d_in[10];
    const float*         lng  = (const float*)d_in[11];
    const float*         lnb  = (const float*)d_in[12];
    const float*         W1   = (const float*)d_in[13];
    const float*         b1   = (const float*)d_in[14];
    const float*         W2   = (const float*)d_in[15];
    const float*         b2   = (const float*)d_in[16];

    float* outx  = (float*)d_out;                        // (M, H) final x
    float* outew = outx + (size_t)M_ * H_;               // (B, E) edge_w

    // workspace: [stats: 2*M floats][msg: M*H floats]  (~214 MB total)
    float* stats = (float*)d_ws;
    float* msg   = stats + (size_t)2 * M_;

    const dim3 blk(256);
    const int mt = M_ / 128;                             // 1616

    // 1) edge weights -> second output section
    edgew_kernel<<<dim3((B_ * E_ + 255) / 256), blk, 0, stream>>>(
        es, ctw, ct, ba, esrc, edst, outew);

    // 2) h_route = elu(X @ proj_W + pb); park in d_out.x (dead after messages)
    gemm_kernel<EPI_ELU, false><<<dim3(mt, H_ / 128), blk, 0, stream>>>(
        bt, nullptr, pW, pb, nullptr, nullptr, outx, H_, H_, 0);

    // 3) messages (LDS scatter-add per batch)
    const int msmem = NB_ * H_ * (int)sizeof(float);     // 103424 B
    (void)hipFuncSetAttribute((const void*)messages_kernel,
                              hipFuncAttributeMaxDynamicSharedMemorySize, msmem);
    messages_kernel<<<dim3(B_), blk, msmem, stream>>>(outx, outew, esrc, edst, msg);

    // 4) x = X + sigmoid([X,msg] @ gate_W + gb) * msg  -> d_out.x
    gemm_kernel<EPI_GATE, true><<<dim3(mt, H_ / 128), blk, 0, stream>>>(
        bt, msg, gW, gb, bt, msg, outx, H_, 2 * H_, H_);

    // 5) two fused pre-norm FFN layers, residual in-place in d_out.x
    (void)hipFuncSetAttribute((const void*)ffn_kernel,
                              hipFuncAttributeMaxDynamicSharedMemorySize, FFN_LDS);
    for (int l = 0; l < 2; ++l) {
        ln_stats_kernel<<<dim3(M_ / 8), blk, 0, stream>>>(outx, stats);
        ffn_kernel<<<dim3(mt), blk, FFN_LDS, stream>>>(
            outx, stats, lng + (size_t)l * H_, lnb + (size_t)l * H_,
            W1 + (size_t)l * H_ * 2 * H_, b1 + (size_t)l * 2 * H_,
            W2 + (size_t)l * 2 * H_ * H_, b2 + (size_t)l * H_,
            outx);
    }
}
